// SelfResidualVQ_75806172775165
// MI455X (gfx1250) — matrix operand order repaired and checked by reading, not compiled, not run on hardware
//
#include <hip/hip_runtime.h>

// ---------------------------------------------------------------------------
// SelfResidualVQ on MI455X (gfx1250, wave32).
//   8 sequential steps; each step = fused GEMM(residual x codebook^T) ->
//   argmin -> gather -> residual update.  GEMM uses v_wmma_f32_16x16x32_bf16
//   with a bf16 hi/lo split (3 WMMAs ~ f32 accuracy).  Codebook and residual
//   are kept as pre-split bf16 hi/lo planes.  Each 64-code codebook chunk is
//   staged in LDS once per block per pass (8 waves share it); WMMA hot loop
//   reads B via ds_load_b128 (hoisted base pointers -> immediate offsets)
//   and A via global b128 (L2-resident).  kc unroll capped at 2 to stay
//   <256 VGPRs.
// ---------------------------------------------------------------------------

typedef __attribute__((ext_vector_type(16))) __bf16 v16bf;
typedef __attribute__((ext_vector_type(8)))  __bf16 v8bf;
typedef __attribute__((ext_vector_type(8)))  float  v8f;
typedef __attribute__((ext_vector_type(4)))  float  v4f;   // native vector for NT stores

#define VQ_D      512
#define VQ_K      1024
#define VQ_Q      8
#define VQ_ROWS   32768L            /* B*N = 8*4096 */
#define WAVES_PB  8
#define ROWS_PW   16
#define ROWS_PB   (WAVES_PB * ROWS_PW)   /* 128 rows per block */
#define NT        4                      /* 4 n-tiles = 64 codes per pass */
#define NC_CODES  (NT * 16)              /* 64 codes staged in LDS per pass */
#define LDS_RS    520                    /* padded LDS row stride (bf16): 1040B
                                            -> start banks {4*lr, 4*lr+4}:
                                            every bank hit exactly 2x per b128 */
#define LDS_PLANE (NC_CODES * LDS_RS)    /* 33280 bf16 per plane */

// float -> bf16 (RNE) hi/lo split ---------------------------------------------
__device__ __forceinline__ void split1(float f, unsigned short& hb16, unsigned short& lb16) {
  unsigned u  = __builtin_bit_cast(unsigned, f);
  unsigned hb = (u + 0x7FFFu + ((u >> 16) & 1u)) >> 16;
  float    fh = __builtin_bit_cast(float, hb << 16);
  float    fl = f - fh;
  unsigned ul = __builtin_bit_cast(unsigned, fl);
  unsigned lb = (ul + 0x7FFFu + ((ul >> 16) & 1u)) >> 16;
  hb16 = (unsigned short)hb;
  lb16 = (unsigned short)lb;
}

// Element-wise split of a plane (codebook once; x before step 0) --------------
__global__ __launch_bounds__(256)
void vq_split_plane(const float* __restrict__ src, __bf16* __restrict__ hi,
                    __bf16* __restrict__ lo, long n) {
  long i = (long)blockIdx.x * 256 + threadIdx.x;
  if (i >= n) return;
  unsigned short hb, lb;
  split1(src[i], hb, lb);
  hi[i] = __builtin_bit_cast(__bf16, hb);
  lo[i] = __builtin_bit_cast(__bf16, lb);
}

// Codebook row norms: one wave per code ---------------------------------------
__global__ __launch_bounds__(256)
void vq_cbsq(const float* __restrict__ cb, float* __restrict__ cbsq) {
  int wave = (blockIdx.x * 256 + threadIdx.x) >> 5;
  int lane = threadIdx.x & 31;
  if (wave >= VQ_K) return;
  const float* row = cb + (long)wave * VQ_D;
  float s = 0.f;
  for (int j = lane; j < VQ_D; j += 32) s += row[j] * row[j];
#pragma unroll
  for (int m = 16; m >= 1; m >>= 1) s += __shfl_xor(s, m, 32);
  if (lane == 0) cbsq[wave] = s;
}

// One RVQ step ----------------------------------------------------------------
__global__ __launch_bounds__(256)
void vq_step(const float*  __restrict__ res_in,    // [ROWS, D] residual f32 (read)
             float*        __restrict__ res_out,   // [ROWS, D] residual f32 (write; may ==res_in)
             const __bf16* __restrict__ ri_hi,     // [ROWS, D] residual hi plane (read)
             const __bf16* __restrict__ ri_lo,     // [ROWS, D] residual lo plane (read)
             __bf16*       __restrict__ ro_hi,     // [ROWS, D] residual hi plane (write)
             __bf16*       __restrict__ ro_lo,     // [ROWS, D] residual lo plane (write)
             const float*  __restrict__ codebook,  // [K, D] f32 (gather phase)
             const __bf16* __restrict__ cb_hi,     // [K, D] codebook hi plane
             const __bf16* __restrict__ cb_lo,     // [K, D] codebook lo plane
             const float*  __restrict__ cb_sq,     // [K] row norms
             float*        __restrict__ quant_out, // [ROWS, D] this step's slice of d_out
             float*        __restrict__ idx_out,   // [ROWS]   indices (as float)
             float*        __restrict__ commit_out,// [1]      commit-loss accumulator
             float inv_count) {
  extern __shared__ __bf16 sB[];          // [2][NC_CODES][LDS_RS] hi | lo planes
  __bf16* sBhi = sB;
  __bf16* sBlo = sB + LDS_PLANE;

  const int lane = threadIdx.x & 31;
  const int wave = threadIdx.x >> 5;
  const int half = lane >> 4;            // K-half for A/B fragments
  const int lr   = lane & 15;
  const long row_base = (long)blockIdx.x * ROWS_PB + (long)wave * ROWS_PW;

  // A-matrix row this lane feeds into WMMA (lanes 0-15 and 16-31 both cover M=0..15)
  const long arow_off = (row_base + lr) * VQ_D;
  const __bf16* ah = ri_hi + arow_off;
  const __bf16* al = ri_lo + arow_off;

  // Per-tile LDS base pointers (invariant across passes; chunk slot is reused).
  // Inner-loop offsets then reduce to 2*kc(+32) bytes -> DS immediate offsets.
  const __bf16* sbh[NT];
  const __bf16* sbl[NT];
#pragma unroll
  for (int t = 0; t < NT; ++t) {
    const int rb = (t * 16 + lr) * LDS_RS + half * 8;
    sbh[t] = sBhi + rb;
    sbl[t] = sBlo + rb;
  }

  float best_d[8];
  int   best_n[8];
#pragma unroll
  for (int v = 0; v < 8; ++v) { best_d[v] = 3.402823466e38f; best_n[v] = 0; }

  const v8f zacc = {0.f, 0.f, 0.f, 0.f, 0.f, 0.f, 0.f, 0.f};

  for (int nc = 0; nc < VQ_K; nc += NC_CODES) {
    // ---- stage 64-code chunk of both codebook planes into LDS (block-wide)
    __syncthreads();   // previous pass finished reading sB
#pragma unroll
    for (int it = 0; it < (NC_CODES * (VQ_D / 16)) / 256; ++it) {   // 8 iters
      const int u    = it * 256 + threadIdx.x;    // 16-element unit
      const int code = u >> 5;                    // u / (512/16)
      const int seg  = u & 31;
      const long goff = (long)(nc + code) * VQ_D + seg * 16;
      const int  loff = code * LDS_RS + seg * 16;
      v16bf gh = *(const v16bf*)(cb_hi + goff);
      v16bf gl = *(const v16bf*)(cb_lo + goff);
      *(v8bf*)(sBhi + loff)     = __builtin_shufflevector(gh, gh, 0, 1, 2, 3, 4, 5, 6, 7);
      *(v8bf*)(sBhi + loff + 8) = __builtin_shufflevector(gh, gh, 8, 9, 10, 11, 12, 13, 14, 15);
      *(v8bf*)(sBlo + loff)     = __builtin_shufflevector(gl, gl, 0, 1, 2, 3, 4, 5, 6, 7);
      *(v8bf*)(sBlo + loff + 8) = __builtin_shufflevector(gl, gl, 8, 9, 10, 11, 12, 13, 14, 15);
    }
    __syncthreads();

    v8f acc[NT];
#pragma unroll
    for (int t = 0; t < NT; ++t) acc[t] = zacc;

#pragma unroll 2      /* cap unroll: keep <256 VGPRs, no vgpr-msb switching */
    for (int kc = 0; kc < VQ_D; kc += 32) {
      // ---- A fragment: two contiguous 8-element bf16 runs per plane (global/L2)
      const int k0 = kc + half * 8;
      v8bf h0 = *(const v8bf*)(ah + k0);
      v8bf h1 = *(const v8bf*)(ah + k0 + 16);
      v8bf l0 = *(const v8bf*)(al + k0);
      v8bf l1 = *(const v8bf*)(al + k0 + 16);
      v16bf ahi = __builtin_shufflevector(h0, h1, 0, 1, 2, 3, 4, 5, 6, 7,
                                          8, 9, 10, 11, 12, 13, 14, 15);
      v16bf alo = __builtin_shufflevector(l0, l1, 0, 1, 2, 3, 4, 5, 6, 7,
                                          8, 9, 10, 11, 12, 13, 14, 15);

#pragma unroll
      for (int t = 0; t < NT; ++t) {
        // ---- B fragment from LDS: base hoisted, kc as immediate offset
        v8bf b0 = *(const v8bf*)(sbh[t] + kc);
        v8bf b1 = *(const v8bf*)(sbh[t] + kc + 16);
        v8bf c0 = *(const v8bf*)(sbl[t] + kc);
        v8bf c1 = *(const v8bf*)(sbl[t] + kc + 16);
        v16bf bhi = __builtin_shufflevector(b0, b1, 0, 1, 2, 3, 4, 5, 6, 7,
                                            8, 9, 10, 11, 12, 13, 14, 15);
        v16bf blo = __builtin_shufflevector(c0, c1, 0, 1, 2, 3, 4, 5, 6, 7,
                                            8, 9, 10, 11, 12, 13, 14, 15);
        // dot ~= hi*hi + hi*lo + lo*hi  (bf16x3 emulated-f32 GEMM)
        acc[t] = __builtin_amdgcn_wmma_f32_16x16x32_bf16(false, ahi, false, bhi,
                                                         (short)0, acc[t], false, false);
        acc[t] = __builtin_amdgcn_wmma_f32_16x16x32_bf16(false, ahi, false, blo,
                                                         (short)0, acc[t], false, false);
        acc[t] = __builtin_amdgcn_wmma_f32_16x16x32_bf16(false, alo, false, bhi,
                                                         (short)0, acc[t], false, false);
      }
    }

    // ---- distances + per-lane argmin update (C layout: VGPR v -> row v / v+8)
#pragma unroll
    for (int t = 0; t < NT; ++t) {
      const int  code = nc + t * 16 + lr;
      const float csq = cb_sq[code];
#pragma unroll
      for (int v = 0; v < 8; ++v) {
        float dist = csq - 2.0f * acc[t][v];
        if (dist < best_d[v] || (dist == best_d[v] && code < best_n[v])) {
          best_d[v] = dist;
          best_n[v] = code;
        }
      }
    }
  }

  // ---- argmin reduction across each 16-lane half (rows 0-7 in half 0, 8-15 in half 1)
#pragma unroll
  for (int v = 0; v < 8; ++v) {
#pragma unroll
    for (int m = 8; m >= 1; m >>= 1) {
      float od = __shfl_xor(best_d[v], m, 32);
      int   on = __shfl_xor(best_n[v], m, 32);
      if (od < best_d[v] || (od == best_d[v] && on < best_n[v])) {
        best_d[v] = od;
        best_n[v] = on;
      }
    }
  }

  // indices out (as float, matching the concatenated f32 output buffer)
  if (lr == 0) {
#pragma unroll
    for (int v = 0; v < 8; ++v)
      idx_out[row_base + half * 8 + v] = (float)best_n[v];
  }

  // broadcast winning index of every row to all 32 lanes
  int idxs[16];
#pragma unroll
  for (int r = 0; r < 16; ++r)
    idxs[r] = __shfl(best_n[r & 7], (r >> 3) << 4, 32);

  // ---- gather + residual update (+ hi/lo re-split) + commit loss
  //      (each lane owns 16 contiguous columns of every row of its wave tile)
  const int col0 = lane * 16;
  float commit = 0.f;
  for (int r = 0; r < 16; ++r) {
    const long roff = (row_base + r) * VQ_D + col0;
    const float* cb = codebook + (long)idxs[r] * VQ_D + col0;
    const float* ri = res_in  + roff;
    float*       ro = res_out + roff;
    float*       qo = quant_out + roff;
    v16bf hv, lv;
#pragma unroll
    for (int j = 0; j < 16; j += 4) {
      v4f qv = *(const v4f*)(cb + j);
      v4f rv = *(const v4f*)(ri + j);
      v4f nr = rv - qv;
      __builtin_nontemporal_store(qv, (v4f*)(qo + j)); // write-only stream
      *(v4f*)(ro + j) = nr;                            // new residual (L2 resident)
      commit += nr.x * nr.x + nr.y * nr.y + nr.z * nr.z + nr.w * nr.w;
      unsigned short hb, lb;
      split1(nr.x, hb, lb);
      hv[j + 0] = __builtin_bit_cast(__bf16, hb); lv[j + 0] = __builtin_bit_cast(__bf16, lb);
      split1(nr.y, hb, lb);
      hv[j + 1] = __builtin_bit_cast(__bf16, hb); lv[j + 1] = __builtin_bit_cast(__bf16, lb);
      split1(nr.z, hb, lb);
      hv[j + 2] = __builtin_bit_cast(__bf16, hb); lv[j + 2] = __builtin_bit_cast(__bf16, lb);
      split1(nr.w, hb, lb);
      hv[j + 3] = __builtin_bit_cast(__bf16, hb); lv[j + 3] = __builtin_bit_cast(__bf16, lb);
    }
    *(v16bf*)(ro_hi + roff) = hv;   // planes for the next step's WMMAs
    *(v16bf*)(ro_lo + roff) = lv;
  }

  // block-reduce commit and atomically accumulate the mean contribution
#pragma unroll
  for (int m = 16; m >= 1; m >>= 1) commit += __shfl_xor(commit, m, 32);
  __shared__ float sc[WAVES_PB];
  if (lane == 0) sc[wave] = commit;
  __syncthreads();
  if (threadIdx.x == 0) {
    float s = 0.f;
#pragma unroll
    for (int w = 0; w < WAVES_PB; ++w) s += sc[w];
    atomicAdd(commit_out, s * inv_count);
  }
}

// -----------------------------------------------------------------------------
extern "C" void kernel_launch(void* const* d_in, const int* in_sizes, int n_in,
                              void* d_out, int out_size, void* d_ws, size_t ws_size,
                              hipStream_t stream) {
  (void)in_sizes; (void)n_in; (void)out_size; (void)ws_size;

  const float* x        = (const float*)d_in[0];   // [8,4096,512] f32
  const float* codebook = (const float*)d_in[1];   // [1024,512]  f32
  // d_in[2] = num_quantizers (fixed at 8 by the reference shapes)

  float* out = (float*)d_out;
  // Output layout (concatenated flat, return order):
  //   quantized [Q, ROWS, D] | indices [Q, ROWS] | commit [Q]
  float* quant_base  = out;
  float* idx_base    = out + (long)VQ_Q * VQ_ROWS * VQ_D;
  float* commit_base = idx_base + (long)VQ_Q * VQ_ROWS;

  // Workspace layout:
  //   residual f32 [ROWS*D]          64 MB
  //   cb_sq [K]                       4 KB
  //   cb_hi, cb_lo [K*D] bf16        1 MB + 1 MB
  //   res_hi, res_lo [ROWS*D] bf16  32 MB + 32 MB
  float*  res    = (float*)d_ws;
  float*  cbsq   = res + VQ_ROWS * VQ_D;
  __bf16* cb_hi  = (__bf16*)(cbsq + VQ_K);
  __bf16* cb_lo  = cb_hi + (long)VQ_K * VQ_D;
  __bf16* res_hi = cb_lo + (long)VQ_K * VQ_D;
  __bf16* res_lo = res_hi + VQ_ROWS * VQ_D;

  (void)hipMemsetAsync(commit_base, 0, VQ_Q * sizeof(float), stream);

  const long NCB = (long)VQ_K * VQ_D;      // 0.5 M elements
  const long NX  = VQ_ROWS * VQ_D;         // 16.8 M elements
  vq_split_plane<<<(int)((NCB + 255) / 256), 256, 0, stream>>>(codebook, cb_hi, cb_lo, NCB);
  vq_split_plane<<<(int)((NX + 255) / 256), 256, 0, stream>>>(x, res_hi, res_lo, NX);
  vq_cbsq<<<(VQ_K * 32 + 255) / 256, 256, 0, stream>>>(codebook, cbsq);

  const size_t lds_bytes = (size_t)2 * LDS_PLANE * sizeof(__bf16);  // 133,120 B
  const float inv_count = 1.0f / (float)(VQ_ROWS * (long)VQ_D);     // mean over B*N*D
  for (int q = 0; q < VQ_Q; ++q) {
    const float* rin = (q == 0) ? x : res;   // never mutate the input x
    vq_step<<<(int)(VQ_ROWS / ROWS_PB), 256, lds_bytes, stream>>>(
        rin, res, res_hi, res_lo, res_hi, res_lo,
        codebook, cb_hi, cb_lo, cbsq,
        quant_base + (long)q * VQ_ROWS * VQ_D,
        idx_base + (long)q * VQ_ROWS,
        commit_base + q,
        inv_count);
  }
}
